// DAHead_Channel_18691697672462
// MI455X (gfx1250) — matrix operand order repaired and checked by reading, not compiled, not run on hardware
//
#include <hip/hip_runtime.h>
#include <hip/hip_bf16.h>

// ---------------------------------------------------------------------------
// DANet channel-attention head for MI455X (gfx1250, wave32)
//   conv1(512->128,3x3) -> BN+ReLU -> channel attention -> conv2(128->512,3x3)
//   -> BN+ReLU
// GEMMs: v_wmma_f32_16x16x32_bf16 (fp32 accum). A-operand tiles staged with
// the Tensor Data Mover (tensor_load_to_lds), double-buffered against WMMA.
// Conv K-dim is tap-major (k = r*Ci + ci) so each 32-wide K-chunk has a
// uniform 3x3 tap: no per-lane div/mod in the im2col inner loop.
// ---------------------------------------------------------------------------

typedef __attribute__((ext_vector_type(16))) __bf16 v16bf;
typedef __attribute__((ext_vector_type(8)))  float  v8f;
typedef __attribute__((ext_vector_type(4)))  unsigned int v4u;
typedef __attribute__((ext_vector_type(8)))  int v8i;
typedef __attribute__((ext_vector_type(4)))  int v4i;

union ABFrag { v16bf v; unsigned int u[8]; };

#define BATCH 8
#define HW    4096
#define PADW  66
#define PADA  (PADW * PADW)   // 4356

__device__ __forceinline__ unsigned short f2bf(float f) {
  unsigned int u = __float_as_uint(f);
  u += 0x7FFFu + ((u >> 16) & 1u);           // round-to-nearest-even
  return (unsigned short)(u >> 16);
}

// --- TDM: 2D bf16 tile (tile0 contiguous elems x tile1 rows) -> LDS --------
// D# per CDNA5 ISA ch.8: group0 = {count/lds_addr/global_addr/type},
// group1 = {data_size, tensor dims, tile dims, dim0 stride}. Groups 2/3 zero
// (tile_dim2 = 0 => 2D). Issued by one wave; tracked with TENSORcnt.
// 6-arg builtin: (uint32x4 g0, int32x8 g1, int32x4, int32x4, int32x8, i32 cpol)
__device__ __forceinline__ void tdm_load_2d_bf16(unsigned ldsAddr,
                                                 const unsigned short* gptr,
                                                 unsigned tile0, unsigned tile1,
                                                 unsigned long long strideElems) {
  unsigned long long ga = (unsigned long long)(uintptr_t)gptr;
  const unsigned td = 0x7FFFFFFFu;           // huge tensor dims: no OOB clip
  v4u g0;
  g0.x = 1u;                                 // count=1, user descriptor
  g0.y = ldsAddr;                            // D#[63:32]  lds_addr (bytes)
  g0.z = (unsigned)ga;                       // D#[95:64]  global_addr[31:0]
  g0.w = (unsigned)((ga >> 32) & 0x01FFFFFFu) | (2u << 30); // addr[56:32], type=2
  v8i g1;
  g1[0] = (int)(1u << 16);                   // wg_mask=0, data_size=1 (2 bytes)
  g1[1] = (int)((td & 0xFFFFu) << 16);       // tensor_dim0[15:0]
  g1[2] = (int)((td >> 16) | ((td & 0xFFFFu) << 16));   // td0 hi | td1 lo
  g1[3] = (int)((td >> 16) | (tile0 << 16)); // td1 hi | tile_dim0
  g1[4] = (int)(tile1 & 0xFFFFu);            // tile_dim1, tile_dim2=0 (2D)
  g1[5] = (int)(unsigned)(strideElems & 0xFFFFFFFFu);   // dim0 stride lo
  g1[6] = (int)(unsigned)(strideElems >> 32);           // stride hi | s1 lo=0
  g1[7] = 0;
  v4i z4 = {0, 0, 0, 0};
  v8i z8 = {0, 0, 0, 0, 0, 0, 0, 0};
  __builtin_amdgcn_tensor_load_to_lds(g0, g1, z4, z4, z8, 0);
}

// --- WMMA fragment loaders (contiguous K order k = 16h + s, wave32) -------------
// A tile in LDS: row-major [rows][16] K-pairs (K=32 chunk).
__device__ __forceinline__ v16bf frag_a(const unsigned int* tile, int mBase, int lane) {
  ABFrag r;
  const unsigned int* row = tile + (mBase + (lane & 15)) * 16;
  const int kh = (lane >> 4) << 3;           // pair offset 0 or 8 (K 0 / 16)
#pragma unroll
  for (int v = 0; v < 4; ++v) { r.u[v] = row[kh + v]; r.u[4 + v] = row[kh + 4 + v]; }
  return r.v;
}
// B tile in LDS: packed pairs [16][Nt]: element (kp, n) = pack(B[2kp][n],B[2kp+1][n])
__device__ __forceinline__ v16bf frag_b(const unsigned int* tile, int nBase, int lane, int Nt) {
  ABFrag r;
  const int n = nBase + (lane & 15);
  const int kp0 = (lane >> 4) << 3;          // pairs 0..7 or 8..15
#pragma unroll
  for (int v = 0; v < 8; ++v) r.u[v] = tile[(kp0 + v) * Nt + n];
  return r.v;
}
// B fragment read *transposed* out of a row-major [rows][16]-pair tile:
// B[k][d] = tile[d][k]  (used for the Gram matrix xf * xf^T)
__device__ __forceinline__ v16bf frag_b_t(const unsigned int* tile, int nBase, int lane) {
  ABFrag r;
  const unsigned int* row = tile + (nBase + (lane & 15)) * 16;
  const int kp0 = (lane >> 4) << 3;
#pragma unroll
  for (int v = 0; v < 8; ++v) r.u[v] = row[kp0 + v];
  return r.v;
}

// ---------------------------------------------------------------------------
// Packing / conversion kernels
// ---------------------------------------------------------------------------
__global__ void zero_u32(unsigned int* __restrict__ p, int n) {
  int i = blockIdx.x * 256 + threadIdx.x;
  if (i < n) p[i] = 0u;
}

// fp32 w[Co][Ci][9] -> bf16 wP[Co][9][Ci]  (tap-major K: k = r*Ci + ci)
__global__ void pack_w_rmajor(const float* __restrict__ w, unsigned short* __restrict__ o,
                              int Ci, int n) {
  int i = blockIdx.x * 256 + threadIdx.x;    // flat over [Co][9][Ci]
  if (i >= n) return;
  int ci = i % Ci;
  int t  = i / Ci;
  int r  = t % 9;
  int co = t / 9;
  o[i] = f2bf(w[(size_t)(co * Ci + ci) * 9 + r]);
}

// [B][C][64][64] fp32 -> [B][C][66][66] bf16 with zero halo
__global__ void pack_pad_bf16(const float* __restrict__ x, unsigned short* __restrict__ xp, int C) {
  long long i = (long long)blockIdx.x * 256 + threadIdx.x;
  long long total = (long long)BATCH * C * PADA;
  if (i >= total) return;
  int xx = (int)(i % PADW);
  int yy = (int)((i / PADW) % PADW);
  long long bc = i / PADA;
  unsigned short v = 0;
  if (yy >= 1 && yy <= 64 && xx >= 1 && xx <= 64)
    v = f2bf(x[bc * HW + (yy - 1) * 64 + (xx - 1)]);
  xp[i] = v;
}

// ---------------------------------------------------------------------------
// 3x3 conv as implicit GEMM with tap-major K:
//   out[b][co][pix] = sum_{r,ci} wP[co][r*Ci+ci] * xPad[b][ci][y+dy][x+dx]
// Block = 256 threads (8 waves). Block tile 128(M) x 128(N), K staged in 32s.
// Weight tile via TDM (double-buffered); im2col B tile: tap uniform per chunk.
// ---------------------------------------------------------------------------
__global__ __launch_bounds__(256) void conv_gemm_bf16(
    const unsigned short* __restrict__ inPad,  // [B][Ci][66][66] bf16
    const unsigned short* __restrict__ wB,     // [Co][9*Ci] bf16 (tap-major)
    float* __restrict__ out,                   // [B][Co][4096] fp32 (raw, pre-BN)
    int Ci, int Co, int ciShift)               // ciShift = log2(Ci) - 5
{
  const int b    = blockIdx.z;
  const int m0   = blockIdx.y * 128;
  const int pix0 = blockIdx.x * 128;
  const int tid  = threadIdx.x;
  const int lane = tid & 31;
  const int wid  = tid >> 5;
  const int wm   = wid >> 2;                  // 0..1
  const int wn   = wid & 3;                   // 0..3
  const int Kt   = Ci * 9;
  const int nK   = Kt >> 5;
  const int ciMask = (1 << ciShift) - 1;

  __shared__ unsigned int As[2][2048];        // double-buffered [128][16] K-pairs
  __shared__ unsigned int Bs[2048];           // [16][128] packed pairs

  const unsigned asAddr0 = (unsigned)(uintptr_t)(void*)&As[0][0];
  const unsigned asAddr1 = (unsigned)(uintptr_t)(void*)&As[1][0];
  const unsigned short* wBase = wB + (size_t)m0 * Kt;
  const unsigned short* xBase = inPad + (size_t)b * Ci * PADA;

  v8f acc[4][2];
#pragma unroll
  for (int i = 0; i < 4; ++i)
#pragma unroll
    for (int j = 0; j < 2; ++j)
#pragma unroll
      for (int e = 0; e < 8; ++e) acc[i][j][e] = 0.f;

  if (wid == 0)  // prologue: DMA first weight tile
    tdm_load_2d_bf16(asAddr0, wBase, 32, 128, (unsigned long long)Kt);

  for (int kc = 0; kc < nK; ++kc) {
    if (wid == 0 && kc + 1 < nK)             // prefetch next tile into other buf
      tdm_load_2d_bf16((kc + 1) & 1 ? asAddr1 : asAddr0,
                       wBase + (kc + 1) * 32, 32, 128, (unsigned long long)Kt);

    // B: im2col tile. Uniform tap for the whole chunk (SALU), channel pair
    // per K-pair -> second element at constant +PADA offset.
    const int rTap   = kc >> ciShift;        // uniform
    const int ciBase = (kc & ciMask) * 32;   // uniform
    const int dy = rTap / 3;
    const int dx = rTap - 3 * dy;
    const int spatU = dy * PADW + dx;        // uniform spatial shift
#pragma unroll
    for (int j = 0; j < 8; ++j) {
      int p = tid + 256 * j;
      int kp = p >> 7, n = p & 127;
      int pix = pix0 + n;
      int y = pix >> 6, x = pix & 63;
      const unsigned short* q =
          xBase + (size_t)(ciBase + kp * 2) * PADA + y * PADW + x + spatU;
      Bs[kp * 128 + n] = (unsigned int)q[0] | ((unsigned int)q[PADA] << 16);
    }
    if (wid == 0) {                          // current tile landed; next may fly
      if (kc + 1 < nK) __builtin_amdgcn_s_wait_tensorcnt(1);
      else             __builtin_amdgcn_s_wait_tensorcnt(0);
    }
    __syncthreads();

    const unsigned int* Ac = As[kc & 1];
    v16bf af[4], bf[2];
#pragma unroll
    for (int mt = 0; mt < 4; ++mt) af[mt] = frag_a(Ac, wm * 64 + mt * 16, lane);
#pragma unroll
    for (int nt = 0; nt < 2; ++nt) bf[nt] = frag_b(Bs, wn * 32 + nt * 16, lane, 128);
#pragma unroll
    for (int mt = 0; mt < 4; ++mt)
#pragma unroll
      for (int nt = 0; nt < 2; ++nt)
        acc[mt][nt] = __builtin_amdgcn_wmma_f32_16x16x32_bf16(
            false, af[mt], false, bf[nt], (short)0, acc[mt][nt], false, false);
    __syncthreads();
  }

  // store D: one base pointer per tile, row stride folds into store offsets
#pragma unroll
  for (int mt = 0; mt < 4; ++mt)
#pragma unroll
    for (int nt = 0; nt < 2; ++nt) {
      int row0 = m0 + wm * 64 + mt * 16 + ((lane >> 4) << 3);
      int col  = pix0 + wn * 32 + nt * 16 + (lane & 15);
      float* p = out + (size_t)(b * Co + row0) * HW + col;
#pragma unroll
      for (int v = 0; v < 8; ++v) p[(size_t)v * HW] = acc[mt][nt][v];
    }
}

// ---------------------------------------------------------------------------
// Per-channel BN statistics: one block per channel, mean + rsqrt(var+eps)
// ---------------------------------------------------------------------------
__global__ void bn_stats(const float* __restrict__ d, int C, float* __restrict__ st) {
  const int c = blockIdx.x, t = threadIdx.x;
  float s = 0.f, q = 0.f;
  for (int i = t; i < BATCH * HW; i += 256) {
    int bb = i >> 12, j = i & (HW - 1);
    float v = d[(size_t)(bb * C + c) * HW + j];
    s += v; q += v * v;
  }
  __shared__ float rs[256], rq[256];
  rs[t] = s; rq[t] = q; __syncthreads();
  for (int k = 128; k > 0; k >>= 1) {
    if (t < k) { rs[t] += rs[t + k]; rq[t] += rq[t + k]; }
    __syncthreads();
  }
  if (t == 0) {
    const float N = (float)(BATCH * HW);
    float mean = rs[0] / N;
    float var  = rq[0] / N - mean * mean;
    st[2 * c]     = mean;
    st[2 * c + 1] = rsqrtf(var + 1e-5f);
  }
}

// BN+ReLU in place; optionally also emit bf16 copy for downstream WMMA
__global__ void bn_relu_apply(float* __restrict__ d, const float* __restrict__ st,
                              const float* __restrict__ gamma, const float* __restrict__ bias,
                              int C, unsigned short* __restrict__ bfout, int n) {
  int i = blockIdx.x * 256 + threadIdx.x;
  if (i >= n) return;
  int c = (i >> 12) % C;
  float y = (d[i] - st[2 * c]) * st[2 * c + 1] * gamma[c] + bias[c];
  y = fmaxf(y, 0.f);
  d[i] = y;
  if (bfout) bfout[i] = f2bf(y);
}

// ---------------------------------------------------------------------------
// Channel Gram matrix: att[b] = hb[b] (128x4096) * hb[b]^T  -> [128][128] fp32
// One TDM-loaded LDS tile per K-chunk serves both A and transposed-B frags.
// ---------------------------------------------------------------------------
__global__ __launch_bounds__(256) void att_gram_bf16(
    const unsigned short* __restrict__ hb,  // [8][128][4096]
    float* __restrict__ att)                // [8][128][128]
{
  const int b = blockIdx.x;
  const int tid = threadIdx.x, lane = tid & 31, wid = tid >> 5;
  const int wm = wid >> 2, wn = wid & 3;

  __shared__ unsigned int Ts[2][2048];       // double-buffered [128 ch][16 pairs]
  const unsigned tsAddr0 = (unsigned)(uintptr_t)(void*)&Ts[0][0];
  const unsigned tsAddr1 = (unsigned)(uintptr_t)(void*)&Ts[1][0];
  const unsigned short* base = hb + (size_t)b * 128 * HW;

  v8f acc[4][2];
#pragma unroll
  for (int i = 0; i < 4; ++i)
#pragma unroll
    for (int j = 0; j < 2; ++j)
#pragma unroll
      for (int e = 0; e < 8; ++e) acc[i][j][e] = 0.f;

  if (wid == 0) tdm_load_2d_bf16(tsAddr0, base, 32, 128, HW);

  for (int kc = 0; kc < 128; ++kc) {
    if (wid == 0) {
      if (kc + 1 < 128) {
        tdm_load_2d_bf16((kc + 1) & 1 ? tsAddr1 : tsAddr0,
                         base + (kc + 1) * 32, 32, 128, HW);
        __builtin_amdgcn_s_wait_tensorcnt(1);
      } else {
        __builtin_amdgcn_s_wait_tensorcnt(0);
      }
    }
    __syncthreads();
    const unsigned int* Tc = Ts[kc & 1];
    v16bf af[4], bf[2];
#pragma unroll
    for (int mt = 0; mt < 4; ++mt) af[mt] = frag_a(Tc, wm * 64 + mt * 16, lane);
#pragma unroll
    for (int nt = 0; nt < 2; ++nt) bf[nt] = frag_b_t(Tc, wn * 32 + nt * 16, lane);
#pragma unroll
    for (int mt = 0; mt < 4; ++mt)
#pragma unroll
      for (int nt = 0; nt < 2; ++nt)
        acc[mt][nt] = __builtin_amdgcn_wmma_f32_16x16x32_bf16(
            false, af[mt], false, bf[nt], (short)0, acc[mt][nt], false, false);
    __syncthreads();
  }
#pragma unroll
  for (int mt = 0; mt < 4; ++mt)
#pragma unroll
    for (int nt = 0; nt < 2; ++nt) {
      int row0 = wm * 64 + mt * 16 + ((lane >> 4) << 3);
      int col  = wn * 32 + nt * 16 + (lane & 15);
      float* p = att + (size_t)(b * 128 + row0) * 128 + col;
#pragma unroll
      for (int v = 0; v < 8; ++v) p[v * 128] = acc[mt][nt][v];
    }
}

// softmax(rowmax - a) == softmax(-a); stable form uses only the row MIN.
__global__ void cam_softmax(const float* __restrict__ att, unsigned short* __restrict__ attB) {
  const int row = blockIdx.x;    // b*128 + c
  const int t = threadIdx.x;     // 128 threads
  __shared__ float red[128];
  float a = att[row * 128 + t];
  red[t] = a; __syncthreads();
  for (int s = 64; s > 0; s >>= 1) { if (t < s) red[t] = fminf(red[t], red[t + s]); __syncthreads(); }
  float mn = red[0]; __syncthreads();
  float e = __expf(mn - a);
  red[t] = e; __syncthreads();
  for (int s = 64; s > 0; s >>= 1) { if (t < s) red[t] += red[t + s]; __syncthreads(); }
  float o = e / red[0];
  attB[row * 128 + t] = f2bf(o);
}

// out = att(128x128) @ hb(128x4096); fused: h2 = beta*out + h, store padded bf16
__global__ __launch_bounds__(256) void cam_out_gemm(
    const unsigned short* __restrict__ attB,  // [8][128][128]
    const unsigned short* __restrict__ hb,    // [8][128][4096]
    const float* __restrict__ h,              // [8][128][4096] fp32 (residual)
    const float* __restrict__ beta,           // [1]
    unsigned short* __restrict__ h2Pad)       // [8][128][66][66]
{
  const int b = blockIdx.z;
  const int pix0 = blockIdx.x * 128;
  const int tid = threadIdx.x, lane = tid & 31, wid = tid >> 5;
  const int wm = wid >> 2, wn = wid & 3;

  __shared__ unsigned int As[2][2048];
  __shared__ unsigned int Bs[2048];
  const unsigned asAddr0 = (unsigned)(uintptr_t)(void*)&As[0][0];
  const unsigned asAddr1 = (unsigned)(uintptr_t)(void*)&As[1][0];
  const unsigned short* aBase = attB + (size_t)b * 128 * 128;

  v8f acc[4][2];
#pragma unroll
  for (int i = 0; i < 4; ++i)
#pragma unroll
    for (int j = 0; j < 2; ++j)
#pragma unroll
      for (int e = 0; e < 8; ++e) acc[i][j][e] = 0.f;

  if (wid == 0) tdm_load_2d_bf16(asAddr0, aBase, 32, 128, 128);

  for (int kc = 0; kc < 4; ++kc) {
    if (wid == 0 && kc + 1 < 4)
      tdm_load_2d_bf16((kc + 1) & 1 ? asAddr1 : asAddr0,
                       aBase + (kc + 1) * 32, 32, 128, 128);
#pragma unroll
    for (int j = 0; j < 8; ++j) {
      int p = tid + 256 * j;
      int kp = p >> 7, n = p & 127;
      int k0 = kc * 32 + kp * 2;
      const unsigned short* q = hb + (size_t)(b * 128 + k0) * HW + pix0 + n;
      Bs[kp * 128 + n] = (unsigned int)q[0] | ((unsigned int)q[HW] << 16);
    }
    if (wid == 0) {
      if (kc + 1 < 4) __builtin_amdgcn_s_wait_tensorcnt(1);
      else            __builtin_amdgcn_s_wait_tensorcnt(0);
    }
    __syncthreads();
    const unsigned int* Ac = As[kc & 1];
    v16bf af[4], bf[2];
#pragma unroll
    for (int mt = 0; mt < 4; ++mt) af[mt] = frag_a(Ac, wm * 64 + mt * 16, lane);
#pragma unroll
    for (int nt = 0; nt < 2; ++nt) bf[nt] = frag_b(Bs, wn * 32 + nt * 16, lane, 128);
#pragma unroll
    for (int mt = 0; mt < 4; ++mt)
#pragma unroll
      for (int nt = 0; nt < 2; ++nt)
        acc[mt][nt] = __builtin_amdgcn_wmma_f32_16x16x32_bf16(
            false, af[mt], false, bf[nt], (short)0, acc[mt][nt], false, false);
    __syncthreads();
  }

  const float bta = beta[0];
#pragma unroll
  for (int mt = 0; mt < 4; ++mt)
#pragma unroll
    for (int nt = 0; nt < 2; ++nt) {
      int row0 = wm * 64 + mt * 16 + ((lane >> 4) << 3);
      int col  = pix0 + wn * 32 + nt * 16 + (lane & 15);
      int y = col >> 6, x = col & 63;
      const float* hp = h + (size_t)(b * 128 + row0) * HW + col;
      unsigned short* op = h2Pad + ((size_t)(b * 128 + row0) * PADW + (y + 1)) * PADW + (x + 1);
#pragma unroll
      for (int v = 0; v < 8; ++v)
        op[(size_t)v * PADA] = f2bf(bta * acc[mt][nt][v] + hp[(size_t)v * HW]);
    }
}

// ---------------------------------------------------------------------------
extern "C" void kernel_launch(void* const* d_in, const int* in_sizes, int n_in,
                              void* d_out, int out_size, void* d_ws, size_t ws_size,
                              hipStream_t stream) {
  (void)in_sizes; (void)n_in; (void)out_size; (void)ws_size;
  const float* x      = (const float*)d_in[0];
  const float* w1     = (const float*)d_in[1];
  const float* gamma1 = (const float*)d_in[2];
  const float* bnb1   = (const float*)d_in[3];
  const float* beta   = (const float*)d_in[4];
  const float* w2     = (const float*)d_in[5];
  const float* gamma2 = (const float*)d_in[6];
  const float* bnb2   = (const float*)d_in[7];
  float* out = (float*)d_out;

  char* ws = (char*)d_ws;
  size_t off = 0;
  auto alloc = [&](size_t bytes) -> void* {
    void* p = ws + off;
    off = (off + bytes + 255) & ~(size_t)255;
    return p;
  };
  unsigned short* xPad  = (unsigned short*)alloc((size_t)BATCH * 512 * PADA * 2); // 35.7 MB
  unsigned short* w1b   = (unsigned short*)alloc((size_t)128 * 4608 * 2);
  unsigned short* w2b   = (unsigned short*)alloc((size_t)512 * 1152 * 2);
  float*          hRaw  = (float*)alloc((size_t)BATCH * 128 * HW * 4);            // 16.8 MB
  unsigned short* hb    = (unsigned short*)alloc((size_t)BATCH * 128 * HW * 2);   // 8.4 MB
  float*          att   = (float*)alloc((size_t)BATCH * 128 * 128 * 4);
  unsigned short* attB  = (unsigned short*)alloc((size_t)BATCH * 128 * 128 * 2);
  unsigned short* h2Pad = (unsigned short*)alloc((size_t)BATCH * 128 * PADA * 2); // 8.9 MB
  float*          st1   = (float*)alloc(128 * 2 * 4);
  float*          st2   = (float*)alloc(512 * 2 * 4);

  // 0) zero h2Pad (halo must be zero every call; interior rewritten below)
  {
    int n = (BATCH * 128 * PADA * 2) / 4;
    zero_u32<<<(n + 255) / 256, 256, 0, stream>>>((unsigned int*)h2Pad, n);
  }
  // 1) pack x (padded bf16) and weights (bf16, tap-major K)
  {
    long long n = (long long)BATCH * 512 * PADA;
    pack_pad_bf16<<<(unsigned)((n + 255) / 256), 256, 0, stream>>>(x, xPad, 512);
    pack_w_rmajor<<<(128 * 4608 + 255) / 256, 256, 0, stream>>>(w1, w1b, 512, 128 * 4608);
    pack_w_rmajor<<<(512 * 1152 + 255) / 256, 256, 0, stream>>>(w2, w2b, 128, 512 * 1152);
  }
  // 2) conv1 implicit GEMM (M=128, K=4608, N=32768); Ci=512 -> ciShift=4
  conv_gemm_bf16<<<dim3(32, 1, BATCH), 256, 0, stream>>>(xPad, w1b, hRaw, 512, 128, 4);
  // 3) BN1 stats + apply (+ bf16 copy)
  bn_stats<<<128, 256, 0, stream>>>(hRaw, 128, st1);
  bn_relu_apply<<<(BATCH * 128 * HW + 255) / 256, 256, 0, stream>>>(
      hRaw, st1, gamma1, bnb1, 128, hb, BATCH * 128 * HW);
  // 4) channel attention: Gram -> softmax(min-stable) -> out GEMM fused residual
  att_gram_bf16<<<BATCH, 256, 0, stream>>>(hb, att);
  cam_softmax<<<BATCH * 128, 128, 0, stream>>>(att, attB);
  cam_out_gemm<<<dim3(32, 1, BATCH), 256, 0, stream>>>(attB, hb, hRaw, beta, h2Pad);
  // 5) conv2 implicit GEMM (M=512, K=1152, N=32768); Ci=128 -> ciShift=2
  conv_gemm_bf16<<<dim3(32, 4, BATCH), 256, 0, stream>>>(h2Pad, w2b, out, 128, 512, 2);
  // 6) BN2 stats + apply in place on d_out
  bn_stats<<<512, 256, 0, stream>>>(out, 512, st2);
  bn_relu_apply<<<(BATCH * 512 * HW + 255) / 256, 256, 0, stream>>>(
      out, st2, gamma2, bnb2, 512, nullptr, BATCH * 512 * HW);
}
